// Attention_73203422593682
// MI455X (gfx1250) — compile-verified, matrix operand order checked by reading, not run on hardware
//
#include <hip/hip_runtime.h>
#include <hip/hip_bf16.h>

typedef __attribute__((ext_vector_type(16))) _Float16 v16h;
typedef __attribute__((ext_vector_type(8)))  _Float16 v8h;
typedef __attribute__((ext_vector_type(4)))  _Float16 v4h;
typedef __attribute__((ext_vector_type(8)))  float    v8f;
typedef int v4i __attribute__((ext_vector_type(4)));

#define BATCH  16
#define SEQ    1024
#define DMODEL 1024
#define NHEAD  16
#define HEADD  64

// ---------------------------------------------------------------------------
// gfx1250 async LDS staging (ASYNCcnt-tracked GLOBAL_LOAD_ASYNC_TO_LDS_B128).
// Probe-discovered signature: (v4i addrspace(1)* src, v4i addrspace(3)* dst,
// imm offset, imm cpol). Guarded so the kernel always compiles.
// ---------------------------------------------------------------------------
#if __has_builtin(__builtin_amdgcn_global_load_async_to_lds_b128) && \
    __has_builtin(__builtin_amdgcn_s_wait_asynccnt)
#define HAVE_ASYNC 1
#else
#define HAVE_ASYNC 0
#endif

typedef __attribute__((address_space(1))) v4i* gptr_v4i;
typedef __attribute__((address_space(3))) v4i* lptr_v4i;

#if HAVE_ASYNC
__device__ __forceinline__ void async_ldg_b128(_Float16* lds_dst, const _Float16* gsrc) {
    __builtin_amdgcn_global_load_async_to_lds_b128(
        (gptr_v4i)gsrc, (lptr_v4i)lds_dst, 0, 0);
}
__device__ __forceinline__ void async_wait0() { __builtin_amdgcn_s_wait_asynccnt(0); }
#else
__device__ __forceinline__ void async_ldg_b128(_Float16* lds_dst, const _Float16* gsrc) {
    *(v8h*)lds_dst = *(const v8h*)gsrc;
}
__device__ __forceinline__ void async_wait0() {}
#endif

// ---------------------------------------------------------------------------
// A/B fragment loader for v_wmma_f32_16x16x32_f16.
// LDS tile is stored N-major (rows) x K-contiguous, ld = K-stride in f16.
// Per ISA 7.12.2 (16-bit A 16x32): lanes 0-15 -> row = lane, K chunks {0..7,16..23};
// lanes 16-31 -> same rows, K chunks {8..15,24..31}. B uses the same striping on Bt.
// ---------------------------------------------------------------------------
__device__ __forceinline__ v16h ld_frag(const _Float16* base, int ld) {
    const int lane = threadIdx.x & 31;
    const int r    = lane & 15;
    const int half = lane >> 4;
    const _Float16* p = base + r * ld + half * 8;
    union { v16h v; v8h h[2]; } u;
    u.h[0] = *(const v8h*)(p);        // K = half*8 + 0..7
    u.h[1] = *(const v8h*)(p + 16);   // K = 16 + half*8 + 0..7
    return u.v;
}

// ---------------------------------------------------------------------------
// Tiled WMMA GEMM:  Out[M,N] = A[M,K] @ W[N,K]^T + bias[N]
// Block tile 128x128, K-slab 64 (two chained wmma K-steps per C tile).
// 256 threads = 8 waves arranged 4 (M) x 2 (N); each wave: 2x4 = 8 C tiles,
// 16 wmma per staging barrier.
// ---------------------------------------------------------------------------
template<bool A_F32, bool OUT_F32>
__global__ __launch_bounds__(256)
void gemm_wmma_kernel(const void* __restrict__ Av, const float* __restrict__ W,
                      const float* __restrict__ bias, void* __restrict__ Outv,
                      int M, int N, int K) {
    __shared__ __align__(16) _Float16 sA[128 * 64];   // 16 KB
    __shared__ __align__(16) _Float16 sB[128 * 64];   // 16 KB

    const int tid  = threadIdx.x;
    const int m0   = blockIdx.y * 128;
    const int n0   = blockIdx.x * 128;
    const int wave = tid >> 5;
    const int wm   = wave & 3;    // 4 row groups of 32
    const int wn   = wave >> 2;   // 2 col groups of 64

    const float*    Af = (const float*)Av;
    const _Float16* Ah = (const _Float16*)Av;

    v8f acc[2][4] = {};

    for (int kt = 0; kt < K; kt += 64) {
        // ---- stage A tile (128 x 64) ----
        if constexpr (A_F32) {
            #pragma unroll
            for (int it = 0; it < 8; ++it) {
                const int idx = tid + it * 256;   // 0..2047 groups of 4
                const int row = idx >> 4;
                const int c4  = (idx & 15) * 4;
                const float* g = Af + (size_t)(m0 + row) * K + kt + c4;
                v4h h; h[0] = (_Float16)g[0]; h[1] = (_Float16)g[1];
                       h[2] = (_Float16)g[2]; h[3] = (_Float16)g[3];
                *(v4h*)&sA[row * 64 + c4] = h;
            }
        } else {
            // raw f16 copy: async LDS staging path (b128 per lane)
            #pragma unroll
            for (int it = 0; it < 4; ++it) {
                const int cidx = tid + it * 256;  // 0..1023 chunks of 8 f16
                const int row  = cidx >> 3;
                const int off  = (cidx & 7) * 8;
                async_ldg_b128(&sA[row * 64 + off],
                               Ah + (size_t)(m0 + row) * K + kt + off);
            }
        }
        // ---- stage B tile from W (fp32 -> f16 convert, must go through VALU) ----
        #pragma unroll
        for (int it = 0; it < 8; ++it) {
            const int idx = tid + it * 256;
            const int row = idx >> 4;
            const int c4  = (idx & 15) * 4;
            const float* gw = W + (size_t)(n0 + row) * K + kt + c4;
            v4h hb; hb[0] = (_Float16)gw[0]; hb[1] = (_Float16)gw[1];
                    hb[2] = (_Float16)gw[2]; hb[3] = (_Float16)gw[3];
            *(v4h*)&sB[row * 64 + c4] = hb;
        }
        // prefetch next K-slab (lowers to global_prefetch_b8 on gfx1250)
        if (kt + 64 < K) {
            const int row = tid >> 1;
            if constexpr (A_F32)
                __builtin_prefetch(Af + (size_t)(m0 + row) * K + kt + 64, 0, 1);
            else
                __builtin_prefetch(Ah + (size_t)(m0 + row) * K + kt + 64, 0, 1);
            __builtin_prefetch(W + (size_t)(n0 + row) * K + kt + 64, 0, 1);
        }
        if constexpr (!A_F32) async_wait0();
        __syncthreads();

        v16h a[2][2], b[4][2];
        #pragma unroll
        for (int i = 0; i < 2; ++i)
            #pragma unroll
            for (int kk = 0; kk < 2; ++kk)
                a[i][kk] = ld_frag(&sA[(wm * 32 + i * 16) * 64 + kk * 32], 64);
        #pragma unroll
        for (int j = 0; j < 4; ++j)
            #pragma unroll
            for (int kk = 0; kk < 2; ++kk)
                b[j][kk] = ld_frag(&sB[(wn * 64 + j * 16) * 64 + kk * 32], 64);
        #pragma unroll
        for (int i = 0; i < 2; ++i)
            #pragma unroll
            for (int j = 0; j < 4; ++j) {
                acc[i][j] = __builtin_amdgcn_wmma_f32_16x16x32_f16(
                    false, a[i][0], false, b[j][0], (short)0, acc[i][j], false, false);
                acc[i][j] = __builtin_amdgcn_wmma_f32_16x16x32_f16(
                    false, a[i][1], false, b[j][1], (short)0, acc[i][j], false, false);
            }
        __syncthreads();
    }

    // epilogue: C layout (ISA 7.12.2): elem e -> row = e + (lane>=16)*8, col = lane&15
    const int lane = tid & 31;
    const int lc   = lane & 15;
    const int lr   = (lane >> 4) * 8;
    #pragma unroll
    for (int i = 0; i < 2; ++i)
        #pragma unroll
        for (int j = 0; j < 4; ++j) {
            const int   col     = n0 + wn * 64 + j * 16 + lc;
            const float bv      = bias[col];
            const int   rowbase = m0 + wm * 32 + i * 16 + lr;
            #pragma unroll
            for (int e = 0; e < 8; ++e) {
                const float  v   = acc[i][j][e] + bv;
                const size_t off = (size_t)(rowbase + e) * N + col;
                if constexpr (OUT_F32) ((float*)Outv)[off] = v;
                else                   ((_Float16*)Outv)[off] = (_Float16)v;
            }
        }
}

// ---------------------------------------------------------------------------
// Flash-attention per (b, h, 64-query tile): online softmax over 64-key tiles.
// S = Q@K^T and O += P@V both via v_wmma_f32_16x16x32_f16.
// Q/K tiles staged with async LDS loads; V is transposed through VALU.
// ---------------------------------------------------------------------------
__global__ __launch_bounds__(256)
void attn_kernel(const _Float16* __restrict__ Q, const _Float16* __restrict__ Kg,
                 const _Float16* __restrict__ Vg, const unsigned char* __restrict__ maskp,
                 _Float16* __restrict__ Og) {
    __shared__ __align__(16) _Float16 sQ [64 * 64];   // queries x hd
    __shared__ __align__(16) _Float16 sKP[64 * 64];   // K tile, then recycled as P
    __shared__ __align__(16) _Float16 sVt[64 * 64];   // V transposed: hd x keys
    __shared__ __align__(16) float    sS [64 * 64];   // raw scores
    __shared__ __align__(16) float    sO [64 * 64];   // output accumulator
    __shared__ float sM[64];
    __shared__ float sL[64];

    const int tid = threadIdx.x;
    const int qt = blockIdx.x, h = blockIdx.y, b = blockIdx.z;
    const int q0 = qt * 64;
    const size_t hoff = (size_t)h * HEADD;

    for (int i = tid; i < 64 * 64; i += 256) sO[i] = 0.0f;
    if (tid < 64) { sM[tid] = -3.0e38f; sL[tid] = 0.0f; }

    {   // stage Q tile (64x64 f16) via async LDS copy
        const int r = tid >> 2, c = (tid & 3) * 16;
        const _Float16* g = Q + (size_t)(b * SEQ + q0 + r) * DMODEL + hoff + c;
        async_ldg_b128(&sQ[r * 64 + c],     g);
        async_ldg_b128(&sQ[r * 64 + c + 8], g + 8);
        async_wait0();
    }
    __syncthreads();

    const int wave = tid >> 5;
    const int lane = tid & 31;
    const int lc   = lane & 15;
    const int lr   = (lane >> 4) * 8;

    for (int kt = 0; kt < SEQ / 64; ++kt) {
        {   // stage K tile (async) and V^T tile (VALU transpose)
            const int r = tid >> 2, c = (tid & 3) * 16;
            const _Float16* gk = Kg + (size_t)(b * SEQ + kt * 64 + r) * DMODEL + hoff + c;
            async_ldg_b128(&sKP[r * 64 + c],     gk);
            async_ldg_b128(&sKP[r * 64 + c + 8], gk + 8);
            const _Float16* gv = Vg + (size_t)(b * SEQ + kt * 64 + r) * DMODEL + hoff + c;
            v8h v0 = *(const v8h*)(gv);
            v8h v1 = *(const v8h*)(gv + 8);
            #pragma unroll
            for (int i = 0; i < 8; ++i) sVt[(c + i) * 64 + r]     = v0[i];
            #pragma unroll
            for (int i = 0; i < 8; ++i) sVt[(c + 8 + i) * 64 + r] = v1[i];
            async_wait0();
        }
        __syncthreads();

        // S = Q @ K^T : 16 tiles of 16x16, 2 per wave; HD=64 => 2 chained wmma
        for (int t = wave; t < 16; t += 8) {
            const int ti = t >> 2, tj = t & 3;
            v16h a0 = ld_frag(&sQ [(ti * 16) * 64], 64);
            v16h a1 = ld_frag(&sQ [(ti * 16) * 64 + 32], 64);
            v16h b0 = ld_frag(&sKP[(tj * 16) * 64], 64);
            v16h b1 = ld_frag(&sKP[(tj * 16) * 64 + 32], 64);
            v8f c = {};
            c = __builtin_amdgcn_wmma_f32_16x16x32_f16(false, a0, false, b0, (short)0, c, false, false);
            c = __builtin_amdgcn_wmma_f32_16x16x32_f16(false, a1, false, b1, (short)0, c, false, false);
            const int  kglob = kt * 64 + tj * 16 + lc;
            const bool mk    = maskp[(size_t)b * SEQ + kglob] != 0;
            #pragma unroll
            for (int e = 0; e < 8; ++e)
                sS[(ti * 16 + lr + e) * 64 + tj * 16 + lc] =
                    mk ? c[e] * 0.125f : -1.0e30f;   // 0.125 = 1/sqrt(64)
        }
        __syncthreads();

        // online softmax: one thread per query row; P overwrites K buffer
        if (tid < 64) {
            const int r = tid;
            float mnew = sM[r];
            for (int k2 = 0; k2 < 64; ++k2) mnew = fmaxf(mnew, sS[r * 64 + k2]);
            const float alpha = __expf(sM[r] - mnew);
            float sum = 0.0f;
            for (int k2 = 0; k2 < 64; ++k2) {
                const float p = __expf(sS[r * 64 + k2] - mnew);
                sKP[r * 64 + k2] = (_Float16)p;
                sum += p;
            }
            sL[r] = sL[r] * alpha + sum;
            sM[r] = mnew;
            for (int k2 = 0; k2 < 64; ++k2) sO[r * 64 + k2] *= alpha;
        }
        __syncthreads();

        // O += P @ V : A = P (q x key), Bt = V^T (hd x key); keys=64 => 2 wmma
        for (int t = wave; t < 16; t += 8) {
            const int ti = t >> 2, tj = t & 3;   // ti: q block, tj: hd block
            v16h a0 = ld_frag(&sKP[(ti * 16) * 64], 64);
            v16h a1 = ld_frag(&sKP[(ti * 16) * 64 + 32], 64);
            v16h b0 = ld_frag(&sVt[(tj * 16) * 64], 64);
            v16h b1 = ld_frag(&sVt[(tj * 16) * 64 + 32], 64);
            v8f c;
            #pragma unroll
            for (int e = 0; e < 8; ++e)
                c[e] = sO[(ti * 16 + lr + e) * 64 + tj * 16 + lc];
            c = __builtin_amdgcn_wmma_f32_16x16x32_f16(false, a0, false, b0, (short)0, c, false, false);
            c = __builtin_amdgcn_wmma_f32_16x16x32_f16(false, a1, false, b1, (short)0, c, false, false);
            #pragma unroll
            for (int e = 0; e < 8; ++e)
                sO[(ti * 16 + lr + e) * 64 + tj * 16 + lc] = c[e];
        }
        __syncthreads();
    }

    {   // normalize and write out (f16)
        const int r = tid >> 2, c = (tid & 3) * 16;
        const float inv = 1.0f / sL[r];
        _Float16* g = Og + (size_t)(b * SEQ + q0 + r) * DMODEL + hoff + c;
        #pragma unroll
        for (int i = 0; i < 16; ++i)
            g[i] = (_Float16)(sO[r * 64 + c + i] * inv);
    }
}

// ---------------------------------------------------------------------------
extern "C" void kernel_launch(void* const* d_in, const int* in_sizes, int n_in,
                              void* d_out, int out_size, void* d_ws, size_t ws_size,
                              hipStream_t stream) {
    (void)in_sizes; (void)n_in; (void)out_size; (void)ws_size;

    const float*         x    = (const float*)d_in[0];
    const unsigned char* mask = (const unsigned char*)d_in[1];
    const float* wq_w = (const float*)d_in[2];
    const float* wq_b = (const float*)d_in[3];
    const float* wk_w = (const float*)d_in[4];
    const float* wk_b = (const float*)d_in[5];
    const float* wv_w = (const float*)d_in[6];
    const float* wv_b = (const float*)d_in[7];
    const float* wo_w = (const float*)d_in[8];
    const float* wo_b = (const float*)d_in[9];

    const int    M   = BATCH * SEQ;            // 16384
    const size_t mat = (size_t)M * DMODEL;     // elems per f16 matrix
    _Float16* Qh = (_Float16*)d_ws;            // 32 MB each
    _Float16* Kh = Qh + mat;
    _Float16* Vh = Kh + mat;
    _Float16* Ah = Vh + mat;

    dim3 gg(DMODEL / 128, M / 128);            // (8, 128)
    gemm_wmma_kernel<true,  false><<<gg, 256, 0, stream>>>(x,  wq_w, wq_b, Qh, M, DMODEL, DMODEL);
    gemm_wmma_kernel<true,  false><<<gg, 256, 0, stream>>>(x,  wk_w, wk_b, Kh, M, DMODEL, DMODEL);
    gemm_wmma_kernel<true,  false><<<gg, 256, 0, stream>>>(x,  wv_w, wv_b, Vh, M, DMODEL, DMODEL);
    attn_kernel<<<dim3(SEQ / 64, NHEAD, BATCH), 256, 0, stream>>>(Qh, Kh, Vh, mask, Ah);
    gemm_wmma_kernel<false, true ><<<gg, 256, 0, stream>>>(Ah, wo_w, wo_b, (float*)d_out, M, DMODEL, DMODEL);
}
